// CascadeMemoryAttention_6777458393275
// MI455X (gfx1250) — matrix operand order repaired and checked by reading, not compiled, not run on hardware
//
#include <hip/hip_runtime.h>
#include <hip/hip_bf16.h>
#include <cstdint>

// ---------------------------------------------------------------------------
// CascadeMemoryAttention for MI455X (gfx1250): bf16 WMMA everywhere matmul-
// shaped, fp32 accumulation, TDM (tensor_load_to_lds) double-buffered weight
// staging in the GEMMs, flash-style online softmax with gated memory keys,
// causal depthwise conv epilogue. Operand loads are batched ahead of WMMA
// runs so a single counter wait covers a dense run of matrix ops.
// ---------------------------------------------------------------------------

typedef __bf16 bf16;
typedef __attribute__((ext_vector_type(16))) __bf16 v16bf;
typedef __attribute__((ext_vector_type(8)))  float  v8f;
typedef __attribute__((ext_vector_type(4)))  unsigned int u32x4;
typedef __attribute__((ext_vector_type(8)))  int i32x8;
typedef __attribute__((ext_vector_type(4)))  int i32x4;

constexpr int Bc = 2;
constexpr int Tc = 2048;
constexpr int Cc = 1024;
constexpr int Hc = 16;
constexpr int Mc = 256;
constexpr int Sc = Tc + 2 * Mc;   // 2560
constexpr int Dc = Cc / Hc;       // 64
constexpr float SCALE = 0.125f;   // 1/sqrt(64)

// POD 16-byte chunk (avoid HIP float4 ctors inside unions)
struct alignas(16) B16x8 { unsigned int w[4]; };

union ABPack {
  v16bf v;
  struct { B16x8 lo; B16x8 hi; } f;
};

union C8 { v8f v; float e[8]; };

__device__ __forceinline__ v8f wmma_bf16(v16bf a, v16bf b, v8f c) {
  // D = A(16x32) * B(32x16) + C, fp32 accumulate
  return __builtin_amdgcn_wmma_f32_16x16x32_bf16(false, a, false, b,
                                                 (short)0, c, false, false);
}

// A-operand (16x32 bf16, M x K): lane holds row = lane&15; elements [0..15] are
// K = hi*16 .. hi*16+15 (matches load_tileB k order)
__device__ __forceinline__ v16bf load_tileA(const bf16* p, int ld) {
  const int lane = threadIdx.x & 31;
  const int row  = lane & 15;
  const int hi   = lane >> 4;
  const bf16* r0 = p + (long)row * ld + hi * 16;
  ABPack u;
  u.f.lo = *(const B16x8*)(r0);
  u.f.hi = *(const B16x8*)(r0 + 8);
  return u.v;
}

// B-operand (32x16 bf16, K x N), sourced from memory where B[n][k] is
// contiguous in k: lane holds col = lane&15, elements are K = kb .. kb+15
// with kb = (lane>=16)*16 (contiguous k order, same as load_tileA).
__device__ __forceinline__ v16bf load_tileB(const bf16* p, int ld) {
  const int lane = threadIdx.x & 31;
  const int col  = lane & 15;
  const int kb   = (lane >> 4) << 4;
  const bf16* r0 = p + (long)col * ld + kb;
  ABPack u;
  u.f.lo = *(const B16x8*)(r0);
  u.f.hi = *(const B16x8*)(r0 + 8);
  return u.v;
}

__device__ __forceinline__ float rowred_max(float v) {
#pragma unroll
  for (int m = 1; m < 16; m <<= 1) v = fmaxf(v, __shfl_xor(v, m, 32));
  return v;
}
__device__ __forceinline__ float rowred_sum(float v) {
#pragma unroll
  for (int m = 1; m < 16; m <<= 1) v += __shfl_xor(v, m, 32);
  return v;
}

// ---------------------------------------------------------------------------
// TDM: 2-D tile load Global -> LDS (Tensor Data Mover, tracked by TENSORcnt)
//   tile_k rows-contiguous elements per row, tile_rows rows, row stride in
//   elements; bf16 (data_size code 1 = 2 bytes). Descriptor per ISA 8.3/8.4.
//   This toolchain's builtin is the 6-arg clang-23 form:
//   (u32x4 g0, i32x8 g1, i32x4 g2, i32x4 g3, i32x8 extra, i32 cpol)
// ---------------------------------------------------------------------------
__device__ __forceinline__ void tdm_load_2d(const bf16* gaddr, bf16* lds_ptr,
                                            unsigned tile_k, unsigned tile_rows,
                                            unsigned row_stride_elems) {
  const unsigned long long ga = (unsigned long long)(uintptr_t)gaddr;
  const unsigned lds_off = (unsigned)(uintptr_t)lds_ptr;  // low 32b = LDS byte addr
  u32x4 g0;
  g0[0] = 1u;                                        // count=1, user D#
  g0[1] = lds_off;                                   // lds_addr
  g0[2] = (unsigned)(ga & 0xFFFFFFFFu);              // global_addr[31:0]
  g0[3] = (unsigned)((ga >> 32) & 0x1FFFFFFu)        // global_addr[56:32]
          | (2u << 30);                              // type = 2 ("image")
  i32x8 g1;
  g1[0] = (int)(1u << 16);                           // data_size=2B; no flags
  g1[1] = (int)((tile_k & 0xFFFFu) << 16);           // tensor_dim0[15:0]
  g1[2] = (int)((tile_k >> 16) | ((tile_rows & 0xFFFFu) << 16));  // td0 hi|td1 lo
  g1[3] = (int)((tile_rows >> 16) | ((tile_k & 0xFFFFu) << 16));  // td1 hi|tile_dim0
  g1[4] = (int)(tile_rows & 0xFFFFu);                // tile_dim1 (tile_dim2=0)
  g1[5] = (int)row_stride_elems;                     // tensor_dim0_stride[31:0]
  g1[6] = 0;                                         // stride0 hi | stride1 lo
  g1[7] = 0;                                         // stride1 hi
  const i32x4 z4 = {0, 0, 0, 0};
  const i32x8 z8 = {0, 0, 0, 0, 0, 0, 0, 0};
  __builtin_amdgcn_tensor_load_to_lds(g0, g1, z4, z4, z8, 0);
}

// ---------------------------------------------------------------------------
// Packing kernels
// ---------------------------------------------------------------------------
__global__ void pack_x_kernel(const float* __restrict__ x,
                              const float* __restrict__ fm,
                              const float* __restrict__ rm,
                              bf16* __restrict__ Xcat) {
  int idx = blockIdx.x * 256 + threadIdx.x;
  if (idx >= Bc * Sc * Cc) return;
  int c = idx % Cc;
  int row = idx / Cc;
  int b = row / Sc;
  int s = row % Sc;
  float v;
  if (s < Tc)            v = x [((long)b * Tc + s) * Cc + c];
  else if (s < Tc + Mc)  v = fm[((long)b * Mc + (s - Tc)) * Cc + c];
  else                   v = rm[((long)b * Mc + (s - Tc - Mc)) * Cc + c];
  Xcat[idx] = (bf16)v;
}

// Wt[n][k] = W[k][n], bf16
__global__ void pack_wt_kernel(const float* __restrict__ W,
                               bf16* __restrict__ Wt) {
  int idx = blockIdx.x * 256 + threadIdx.x;
  if (idx >= Cc * Cc) return;
  int kk = idx % Cc;
  int n  = idx / Cc;
  Wt[idx] = (bf16)W[(long)kk * Cc + n];
}

// ---------------------------------------------------------------------------
// Generic bf16 WMMA GEMM: Y[m][n] = sum_k A[m][k] * Wt[n][k]
//   block = 256 threads = 8 waves; wave computes 16 rows x 64 cols; the block
//   shares one 64x64 bf16 weight panel staged in LDS by the TDM (wave 0
//   issues tensor_load_to_lds; double buffered; s_wait_tensorcnt + barrier
//   publishes each panel; one barrier/step is WAR-safe because arrival
//   implies the previous step's WMMAs -- and their DS reads -- completed).
//   All 8 B-tiles + both A-tiles are loaded before the 8-WMMA run so one
//   counter wait covers the whole matrix-op burst.
// mode 0: store bf16 row-major (ld = Cc)
// mode 2: store bf16 into per-head transposed V layout Vt[b][h][d][S]
// mode 3: store fp32 row-major (final output)
// ---------------------------------------------------------------------------
__global__ __launch_bounds__(256) void gemm_bf16_kernel(
    const bf16* __restrict__ A, const bf16* __restrict__ Bt,
    int rows_out_per_b, int rows_in_per_b, int Mrows, int mode,
    bf16* __restrict__ outb, float* __restrict__ outf) {
  __shared__ __align__(16) bf16 Bpanel[2][64 * 64];   // 2 x 8KB

  const int wave = threadIdx.x >> 5;
  const int lane = threadIdx.x & 31;
  const int row0 = blockIdx.x * 128 + wave * 16;      // grids are exact
  const int col0 = blockIdx.y * 64;

  const int b   = row0 / rows_out_per_b;
  const int tin = row0 % rows_out_per_b;
  const bf16* Abase = A + ((long)b * rows_in_per_b + tin) * Cc;
  const bf16* Btile = Bt + (long)col0 * Cc;           // 64 rows x Cc, row-major

  C8 acc[4];
#pragma unroll
  for (int n = 0; n < 4; n++)
#pragma unroll
    for (int i = 0; i < 8; i++) acc[n].e[i] = 0.0f;

  constexpr int KSTEPS = Cc / 64;                     // 16 TDM panels
  if (wave == 0)
    tdm_load_2d(Btile, &Bpanel[0][0], 64, 64, Cc);

  for (int step = 0; step < KSTEPS; ++step) {
    if (wave == 0) __builtin_amdgcn_s_wait_tensorcnt(0);
    __syncthreads();                                  // publish panel step&1
    if (wave == 0 && step + 1 < KSTEPS)
      tdm_load_2d(Btile + (step + 1) * 64, &Bpanel[(step + 1) & 1][0], 64, 64, Cc);

    const bf16* bp = &Bpanel[step & 1][0];
    __builtin_prefetch(Abase + (step + 1) * 64);

    // batch all operand loads, then a dense run of 8 WMMAs
    const v16bf a0 = load_tileA(Abase + step * 64, Cc);
    const v16bf a1 = load_tileA(Abase + step * 64 + 32, Cc);
    v16bf bt[8];
#pragma unroll
    for (int n = 0; n < 4; n++) {
      bt[n]     = load_tileB(bp + n * 16 * 64, 64);
      bt[4 + n] = load_tileB(bp + n * 16 * 64 + 32, 64);
    }
#pragma unroll
    for (int n = 0; n < 4; n++) acc[n].v = wmma_bf16(a0, bt[n], acc[n].v);
#pragma unroll
    for (int n = 0; n < 4; n++) acc[n].v = wmma_bf16(a1, bt[4 + n], acc[n].v);
  }

  const int hi = lane >> 4;
  const int cl = lane & 15;
#pragma unroll
  for (int n = 0; n < 4; n++) {
#pragma unroll
    for (int r = 0; r < 8; r++) {
      const int orow = row0 + hi * 8 + r;
      const int ocol = col0 + n * 16 + cl;
      const float val = acc[n].e[r];
      if (mode == 0) {
        outb[(long)orow * Cc + ocol] = (bf16)val;
      } else if (mode == 2) {
        const int b2 = orow / Sc, s = orow % Sc;
        const int h = ocol / Dc, dd = ocol % Dc;
        outb[(((long)(b2 * Hc + h) * Dc) + dd) * Sc + s] = (bf16)val;
      } else {
        outf[(long)orow * Cc + ocol] = val;
      }
    }
  }
}

// ---------------------------------------------------------------------------
// Gate: g[b,h,t] = sigmoid(q[b,t,:] . gate_W[:,h] + gate_b[h])
// one wave per token
// ---------------------------------------------------------------------------
__global__ __launch_bounds__(256) void gate_kernel(
    const bf16* __restrict__ Qb, const float* __restrict__ gW,
    const float* __restrict__ gb, float* __restrict__ gate) {
  const int wave = threadIdx.x >> 5;
  const int lane = threadIdx.x & 31;
  const int tok = blockIdx.x * 8 + wave;
  if (tok >= Bc * Tc) return;
  const int b = tok / Tc, t = tok % Tc;

  float acc[Hc];
#pragma unroll
  for (int h = 0; h < Hc; h++) acc[h] = 0.0f;

  for (int i = lane; i < Cc; i += 32) {
    const float qv = (float)Qb[(long)tok * Cc + i];
    const float* g = gW + (long)i * Hc;
#pragma unroll
    for (int h = 0; h < Hc; h++) acc[h] += qv * g[h];
  }
#pragma unroll
  for (int h = 0; h < Hc; h++) {
    float v = acc[h];
#pragma unroll
    for (int m = 1; m < 32; m <<= 1) v += __shfl_xor(v, m, 32);
    if (lane == 0)
      gate[((long)(b * Hc + h)) * Tc + t] = 1.0f / (1.0f + __expf(-(v + gb[h])));
  }
}

// ---------------------------------------------------------------------------
// Flash attention with gated memory keys.
// one wave per 16-query tile of one (b,h); 8 waves per block.
// dual accumulators (chunk keys / memory keys) share one online softmax state.
// ---------------------------------------------------------------------------
__global__ __launch_bounds__(256) void attn_kernel(
    const bf16* __restrict__ Qb, const bf16* __restrict__ Kb,
    const bf16* __restrict__ Vt, const float* __restrict__ gate,
    float* __restrict__ Y) {
  __shared__ __align__(16) bf16 Pl[8][16][40];   // P-tile staging, per wave

  const int wave = threadIdx.x >> 5;
  const int lane = threadIdx.x & 31;
  const int gt = blockIdx.x * 8 + wave;          // global 16-row tile id
  const int qt = gt & 127;                       // T/16 = 128 tiles per (b,h)
  const int bh = gt >> 7;
  const int b = bh / Hc, h = bh % Hc;
  const int q0 = qt * 16;
  const int hi = lane >> 4;
  const int cl = lane & 15;

  // Q A-operand tiles (d = 64 -> two K=32 steps), loaded once
  const bf16* qbase = Qb + ((long)(b * Tc + q0)) * Cc + h * Dc;
  const v16bf qa0 = load_tileA(qbase, Cc);
  const v16bf qa1 = load_tileA(qbase + 32, Cc);

  C8 accc[4], accm[4];
  float mrow[8], lrow[8];
#pragma unroll
  for (int n = 0; n < 4; n++)
#pragma unroll
    for (int i = 0; i < 8; i++) { accc[n].e[i] = 0.0f; accm[n].e[i] = 0.0f; }
#pragma unroll
  for (int r = 0; r < 8; r++) { mrow[r] = -1e30f; lrow[r] = 0.0f; }

  bf16 (*P)[40] = Pl[wave];

  auto process = [&](int sblk, bool causal, C8* acc) {
    // scores: 16 queries x 32 keys = two 16x16 C tiles, K = d = 64
    C8 sc0, sc1;
#pragma unroll
    for (int i = 0; i < 8; i++) { sc0.e[i] = 0.0f; sc1.e[i] = 0.0f; }
    const bf16* kbase = Kb + ((long)(b * Sc + sblk)) * Cc + h * Dc;
    {
      v16bf b00 = load_tileB(kbase, Cc);
      v16bf b01 = load_tileB(kbase + 32, Cc);
      v16bf b10 = load_tileB(kbase + (long)16 * Cc, Cc);
      v16bf b11 = load_tileB(kbase + (long)16 * Cc + 32, Cc);
      sc0.v = wmma_bf16(qa0, b00, sc0.v);
      sc0.v = wmma_bf16(qa1, b01, sc0.v);
      sc1.v = wmma_bf16(qa0, b10, sc1.v);
      sc1.v = wmma_bf16(qa1, b11, sc1.v);
    }
    // scale + causal mask (finite -inf to avoid inf-inf NaN)
#pragma unroll
    for (int r = 0; r < 8; r++) {
      sc0.e[r] *= SCALE;
      sc1.e[r] *= SCALE;
      if (causal) {
        const int rowq = q0 + hi * 8 + r;
        if (sblk + cl > rowq)      sc0.e[r] = -1e30f;
        if (sblk + 16 + cl > rowq) sc1.e[r] = -1e30f;
      }
    }
    // online softmax (rows live in half-wave lane groups)
#pragma unroll
    for (int r = 0; r < 8; r++) {
      float mx = rowred_max(fmaxf(sc0.e[r], sc1.e[r]));
      const float mn = fmaxf(mrow[r], mx);
      const float rescale = __expf(mrow[r] - mn);
      mrow[r] = mn;
      const float p0 = __expf(sc0.e[r] - mn);
      const float p1 = __expf(sc1.e[r] - mn);
      sc0.e[r] = p0;
      sc1.e[r] = p1;
      lrow[r] = lrow[r] * rescale + rowred_sum(p0 + p1);
#pragma unroll
      for (int n = 0; n < 4; n++) {
        accc[n].e[r] *= rescale;   // both accumulators share the denominator
        accm[n].e[r] *= rescale;
      }
    }
    // P tile (C layout) -> LDS -> A layout, bf16
#pragma unroll
    for (int r = 0; r < 8; r++) {
      P[hi * 8 + r][cl]      = (bf16)sc0.e[r];
      P[hi * 8 + r][16 + cl] = (bf16)sc1.e[r];
    }
    asm volatile("s_wait_dscnt 0" ::: "memory");  // wave-private LDS region
    const v16bf pa = load_tileA(&P[0][0], 40);
    // PV: P(16x32) x V(32x64) via 4 N-tiles; V stored transposed per head.
    // Batch the 4 V-tile loads, then a dense 4-WMMA run.
    const bf16* vbase = Vt + ((long)(b * Hc + h) * Dc) * Sc + sblk;
    v16bf vb[4];
#pragma unroll
    for (int n = 0; n < 4; n++)
      vb[n] = load_tileB(vbase + (long)(n * 16) * Sc, Sc);
#pragma unroll
    for (int n = 0; n < 4; n++)
      acc[n].v = wmma_bf16(pa, vb[n], acc[n].v);
  };

  // chunk keys (causal)
  for (int sblk = 0; sblk < q0 + 16; sblk += 32) process(sblk, true, accc);
  // memory keys (unmasked, gated)
  for (int sblk = Tc; sblk < Tc + 2 * Mc; sblk += 32) process(sblk, false, accm);

  // epilogue: y = (acc_chunk + g * acc_mem) / l
#pragma unroll
  for (int r = 0; r < 8; r++) {
    const int rowq = q0 + hi * 8 + r;
    const float g = gate[((long)(b * Hc + h)) * Tc + rowq];
    const float inv = 1.0f / lrow[r];
#pragma unroll
    for (int n = 0; n < 4; n++) {
      const float y = (accc[n].e[r] + g * accm[n].e[r]) * inv;
      Y[((long)(b * Tc + rowq)) * Cc + h * Dc + n * 16 + cl] = y;
    }
  }
}

// ---------------------------------------------------------------------------
// Causal depthwise conv1d (K=4) + bias + residual; emit bf16 for final GEMM.
// ---------------------------------------------------------------------------
__global__ void canon_kernel(const float* __restrict__ Y,
                             const float* __restrict__ cw,
                             const float* __restrict__ cb,
                             bf16* __restrict__ y2) {
  int idx = blockIdx.x * 256 + threadIdx.x;
  if (idx >= Bc * Tc * Cc) return;
  int c = idx % Cc;
  int row = idx / Cc;
  int t = row % Tc;
  int b = row / Tc;
  float acc = cb[c];
#pragma unroll
  for (int j = 0; j < 4; j++) {
    const int tt = t - 3 + j;
    if (tt >= 0) acc += cw[c * 4 + j] * Y[((long)(b * Tc + tt)) * Cc + c];
  }
  y2[idx] = (bf16)(Y[idx] + acc);
}

// ---------------------------------------------------------------------------
// Host-side launch
// ---------------------------------------------------------------------------
extern "C" void kernel_launch(void* const* d_in, const int* in_sizes, int n_in,
                              void* d_out, int out_size, void* d_ws,
                              size_t ws_size, hipStream_t stream) {
  (void)in_sizes; (void)n_in; (void)out_size; (void)ws_size;
  const float* x  = (const float*)d_in[0];
  const float* fm = (const float*)d_in[1];
  const float* rm = (const float*)d_in[2];
  const float* Wq = (const float*)d_in[3];
  const float* Wk = (const float*)d_in[4];
  const float* Wv = (const float*)d_in[5];
  const float* Wo = (const float*)d_in[6];
  const float* gW = (const float*)d_in[7];
  const float* gb = (const float*)d_in[8];
  const float* cw = (const float*)d_in[9];
  const float* cb = (const float*)d_in[10];
  float* out = (float*)d_out;

  char* p = (char*)d_ws;
  auto alloc_bf = [&](size_t n) {
    bf16* r = (bf16*)p; p += ((n * sizeof(bf16) + 255) / 256) * 256; return r; };
  auto alloc_f = [&](size_t n) {
    float* r = (float*)p; p += ((n * sizeof(float) + 255) / 256) * 256; return r; };

  bf16* Xcat = alloc_bf((size_t)Bc * Sc * Cc);
  bf16* WqT  = alloc_bf((size_t)Cc * Cc);
  bf16* WkT  = alloc_bf((size_t)Cc * Cc);
  bf16* WvT  = alloc_bf((size_t)Cc * Cc);
  bf16* WoT  = alloc_bf((size_t)Cc * Cc);
  bf16* Qb   = alloc_bf((size_t)Bc * Tc * Cc);
  bf16* Kb   = alloc_bf((size_t)Bc * Sc * Cc);
  bf16* Vt   = alloc_bf((size_t)Bc * Sc * Cc);
  bf16* y2b  = alloc_bf((size_t)Bc * Tc * Cc);
  float* gatev = alloc_f((size_t)Bc * Hc * Tc);
  float* Yf    = alloc_f((size_t)Bc * Tc * Cc);

  // 1) pack inputs / weights to bf16 (weights transposed for B-operand)
  pack_x_kernel<<<(Bc * Sc * Cc + 255) / 256, 256, 0, stream>>>(x, fm, rm, Xcat);
  pack_wt_kernel<<<(Cc * Cc + 255) / 256, 256, 0, stream>>>(Wq, WqT);
  pack_wt_kernel<<<(Cc * Cc + 255) / 256, 256, 0, stream>>>(Wk, WkT);
  pack_wt_kernel<<<(Cc * Cc + 255) / 256, 256, 0, stream>>>(Wv, WvT);
  pack_wt_kernel<<<(Cc * Cc + 255) / 256, 256, 0, stream>>>(Wo, WoT);

  // 2) projections (WMMA + TDM-staged weights)
  gemm_bf16_kernel<<<dim3((Bc * Tc) / 128, Cc / 64), 256, 0, stream>>>(
      Xcat, WqT, Tc, Sc, Bc * Tc, 0, Qb, nullptr);
  gemm_bf16_kernel<<<dim3((Bc * Sc) / 128, Cc / 64), 256, 0, stream>>>(
      Xcat, WkT, Sc, Sc, Bc * Sc, 0, Kb, nullptr);
  gemm_bf16_kernel<<<dim3((Bc * Sc) / 128, Cc / 64), 256, 0, stream>>>(
      Xcat, WvT, Sc, Sc, Bc * Sc, 2, Vt, nullptr);

  // 3) gate
  gate_kernel<<<(Bc * Tc) / 8, 256, 0, stream>>>(Qb, gW, gb, gatev);

  // 4) gated flash attention (WMMA)
  attn_kernel<<<(Bc * Hc * (Tc / 16)) / 8, 256, 0, stream>>>(Qb, Kb, Vt, gatev, Yf);

  // 5) causal depthwise conv + residual -> bf16
  canon_kernel<<<(Bc * Tc * Cc + 255) / 256, 256, 0, stream>>>(Yf, cw, cb, y2b);

  // 6) output projection (WMMA + TDM-staged weights) -> fp32 d_out
  gemm_bf16_kernel<<<dim3((Bc * Tc) / 128, Cc / 64), 256, 0, stream>>>(
      y2b, WoT, Tc, Tc, Bc * Tc, 3, nullptr, out);
}
